// TripletLoss_33509334843814
// MI455X (gfx1250) — matrix operand order repaired and checked by reading, not compiled, not run on hardware
//
#include <hip/hip_runtime.h>
#include <hip/hip_bf16.h>

typedef __attribute__((ext_vector_type(16))) __bf16 v16bf_t;
typedef __attribute__((ext_vector_type(8)))  __bf16 v8bf_t;
typedef __attribute__((ext_vector_type(4)))  __bf16 v4bf_t;
typedef __attribute__((ext_vector_type(8)))  float  v8f_t;

#define MARGIN_F 1.0f
#define EPS_F    1e-6f
#define INF_F    1e9f

// ---------------------------------------------------------------------------
// helpers
// ---------------------------------------------------------------------------
__device__ __forceinline__ v8f_t wmma_bf16(v16bf_t a, v16bf_t b, v8f_t c) {
  // (neg_a, A, neg_b, B, c_mod, C, reuse_a, reuse_b)
  return __builtin_amdgcn_wmma_f32_16x16x32_bf16(false, a, false, b, (short)0, c,
                                                 false, false);
}

__device__ __forceinline__ v16bf_t cat8(v8bf_t a, v8bf_t b) {
  return __builtin_shufflevector(a, b, 0, 1, 2, 3, 4, 5, 6, 7,
                                 8, 9, 10, 11, 12, 13, 14, 15);
}

// split 16 f32 into bf16 hi + bf16 lo (x ~= hi + lo, |lo| <~ 2^-8 |x|)
__device__ __forceinline__ void cvt_split16(const float* __restrict__ f,
                                            v16bf_t& hi, v16bf_t& lo) {
#pragma unroll
  for (int t = 0; t < 16; ++t) {
    float x  = f[t];
    __bf16 hb = (__bf16)x;
    float  hf = (float)hb;
    hi[t] = hb;
    lo[t] = (__bf16)(x - hf);
  }
}

// fallback-path raw f32 tile loads
__device__ __forceinline__ void ldA16(const float* __restrict__ p,
                                      float* __restrict__ d) {
  *(float4*)(d + 0)  = *(const float4*)(p + 0);
  *(float4*)(d + 4)  = *(const float4*)(p + 4);
  *(float4*)(d + 8)  = *(const float4*)(p + 8);
  *(float4*)(d + 12) = *(const float4*)(p + 12);
}
__device__ __forceinline__ void ldB16(const float* __restrict__ p,
                                      float* __restrict__ d) {
  *(float4*)(d + 0)  = *(const float4*)(p + 0);
  *(float4*)(d + 4)  = *(const float4*)(p + 4);
  *(float4*)(d + 8)  = *(const float4*)(p + 8);
  *(float4*)(d + 12) = *(const float4*)(p + 12);
}

// ---------------------------------------------------------------------------
// kernel 0a: one-shot bf16 hi/lo split of E (fast path)
// ---------------------------------------------------------------------------
__global__ __launch_bounds__(256) void split_kernel(const float* __restrict__ E,
                                                    __bf16* __restrict__ hi,
                                                    __bf16* __restrict__ lo,
                                                    int total4) {
  const int i = blockIdx.x * 256 + threadIdx.x;
  if (i >= total4) return;
  const float4 x = ((const float4*)E)[i];
  float xs[4] = {x.x, x.y, x.z, x.w};
  v4bf_t hv, lv;
#pragma unroll
  for (int t = 0; t < 4; ++t) {
    __bf16 hb = (__bf16)xs[t];
    hv[t] = hb;
    lv[t] = (__bf16)(xs[t] - (float)hb);
  }
  *(v4bf_t*)(hi + 4 * (size_t)i) = hv;
  *(v4bf_t*)(lo + 4 * (size_t)i) = lv;
}

// ---------------------------------------------------------------------------
// kernel 0b: row squared norms
// ---------------------------------------------------------------------------
__global__ __launch_bounds__(256) void sq_kernel(const float* __restrict__ E,
                                                 float* __restrict__ sq,
                                                 int N, int D) {
  const int i = blockIdx.x;
  const float* row = E + (size_t)i * D;
  float s = 0.f;
  for (int d = threadIdx.x; d < D; d += 256) {
    float x = row[d];
    s += x * x;
  }
#pragma unroll
  for (int off = 16; off; off >>= 1) s += __shfl_xor(s, off, 32);
  __shared__ float p[8];
  if ((threadIdx.x & 31) == 0) p[threadIdx.x >> 5] = s;
  __syncthreads();
  if (threadIdx.x == 0) {
    float t = 0.f;
    for (int j = 0; j < 8; ++j) t += p[j];
    sq[i] = t;
  }
}

// ---------------------------------------------------------------------------
// mining pass. Block = 256 thr (8 waves) owns 16 anchor rows; each wave walks
// column-tile pairs (2 x 16 cols) reusing its A operand. bf16x3 WMMA Gram.
// PRE:  operands come from pre-split Ehi/Elo (bf16 loads, no inner-loop VALU).
// !PRE: operands converted on the fly from f32 E (ws too small for split).
// PASS 0: d_p (max over positives) + pos_idx + hardest-neg idx.
// PASS 1: semi-hard neg (min dist in (d_p, d_p+margin)), fallback hardest.
// ---------------------------------------------------------------------------
template <int PASS, bool PRE>
__global__ __launch_bounds__(256) void mine_pass(
    const float* __restrict__ E, const __bf16* __restrict__ Ehi,
    const __bf16* __restrict__ Elo, const int* __restrict__ cls,
    const float* __restrict__ sq, const float* __restrict__ dp_in,
    const int* __restrict__ negh_in, float* __restrict__ dp_out,
    int* __restrict__ posi_out, int* __restrict__ negh_out,
    int* __restrict__ negsel_out, int N, int D) {
  const int tid  = threadIdx.x;
  const int w    = tid >> 5;
  const int lane = tid & 31;
  const int nl   = lane & 15;   // row-in-tile (A) / col-in-tile (B,C)
  const int h    = lane >> 4;   // lane half
  const int rb   = blockIdx.x * 16;

  // row-side constants: this lane's C rows are {v + 8h}
  float sqr[8];
  int   clr[8];
  float dpr[8];
#pragma unroll
  for (int v = 0; v < 8; ++v) {
    const int r = rb + v + 8 * h;
    sqr[v] = sq[r];
    clr[v] = cls[r];
    dpr[v] = (PASS == 1) ? dp_in[r] : 0.f;
  }

  float bPd[8], bNd[8];
  int   bPi[8], bNi[8];
#pragma unroll
  for (int v = 0; v < 8; ++v) {
    bPd[v] = -INF_F; bPi[v] = 0;   // pos: lexicographic max (d, -idx)
    bNd[v] =  INF_F; bNi[v] = 0;   // neg/semi: lexicographic min (d, idx)
  }

  const size_t rowA = (size_t)(rb + nl) * D;
  const int npair = N >> 5;  // column tile-pairs

  for (int pr = w; pr < npair; pr += 8) {
    const int cb0 = pr * 32;
    const int cb1 = cb0 + 16;
    const size_t rowB0 = (size_t)(cb0 + nl) * D;
    const size_t rowB1 = (size_t)(cb1 + nl) * D;

    v8f_t c0 = {0.f, 0.f, 0.f, 0.f, 0.f, 0.f, 0.f, 0.f};
    v8f_t c1 = {0.f, 0.f, 0.f, 0.f, 0.f, 0.f, 0.f, 0.f};

    for (int k0 = 0; k0 < D; k0 += 32) {
      v16bf_t ah, al, b0h, b0l, b1h, b1l;
      if constexpr (PRE) {
        const size_t oa = rowA + k0 + 16 * h;    // A: 16 consecutive K
        ah = cat8(*(const v8bf_t*)(Ehi + oa), *(const v8bf_t*)(Ehi + oa + 8));
        al = cat8(*(const v8bf_t*)(Elo + oa), *(const v8bf_t*)(Elo + oa + 8));
        const size_t o0 = rowB0 + k0 + 16 * h;   // B: 16 consecutive K
        b0h = cat8(*(const v8bf_t*)(Ehi + o0), *(const v8bf_t*)(Ehi + o0 + 8));
        b0l = cat8(*(const v8bf_t*)(Elo + o0), *(const v8bf_t*)(Elo + o0 + 8));
        const size_t o1 = rowB1 + k0 + 16 * h;
        b1h = cat8(*(const v8bf_t*)(Ehi + o1), *(const v8bf_t*)(Ehi + o1 + 8));
        b1l = cat8(*(const v8bf_t*)(Elo + o1), *(const v8bf_t*)(Elo + o1 + 8));
      } else {
        float af[16], b0f[16], b1f[16];
        ldA16(E + rowA + k0 + 16 * h, af);
        ldB16(E + rowB0 + k0 + 16 * h, b0f);
        ldB16(E + rowB1 + k0 + 16 * h, b1f);
        cvt_split16(af,  ah,  al);
        cvt_split16(b0f, b0h, b0l);
        cvt_split16(b1f, b1h, b1l);
      }
      // G ~= Ahi*Bhi + Ahi*Blo + Alo*Bhi  (bf16x3, f32 accumulate)
      c0 = wmma_bf16(ah, b0h, c0);
      c0 = wmma_bf16(ah, b0l, c0);
      c0 = wmma_bf16(al, b0h, c0);
      c1 = wmma_bf16(ah, b1h, c1);
      c1 = wmma_bf16(ah, b1l, c1);
      c1 = wmma_bf16(al, b1h, c1);
    }

#pragma unroll
    for (int tile = 0; tile < 2; ++tile) {
      const int cb = tile ? cb1 : cb0;
      const int col = cb + nl;
      const int   cc  = cls[col];
      const float sqc = sq[col];
#pragma unroll
      for (int v = 0; v < 8; ++v) {
        const float g  = tile ? c1[v] : c0[v];
        const float d2 = sqr[v] + sqc - 2.f * g;
        const float d  = sqrtf(fmaxf(d2, 0.f));
        const int row  = rb + v + 8 * h;
        const bool same = (clr[v] == cc);
        if (PASS == 0) {
          if (same && (row != col)) {
            if (d > bPd[v] || (d == bPd[v] && col < bPi[v])) { bPd[v] = d; bPi[v] = col; }
          }
          if (!same) {
            if (d < bNd[v] || (d == bNd[v] && col < bNi[v])) { bNd[v] = d; bNi[v] = col; }
          }
        } else {
          if (!same && d > dpr[v] && d < dpr[v] + MARGIN_F) {
            if (d < bNd[v] || (d == bNd[v] && col < bNi[v])) { bNd[v] = d; bNi[v] = col; }
          }
        }
      }
    }
  }

  // cross-lane butterfly within each 16-lane half (rows identical per half)
#pragma unroll
  for (int v = 0; v < 8; ++v) {
#pragma unroll
    for (int off = 1; off < 16; off <<= 1) {
      if (PASS == 0) {
        float od = __shfl_xor(bPd[v], off, 32);
        int   oi = __shfl_xor(bPi[v], off, 32);
        if (od > bPd[v] || (od == bPd[v] && oi < bPi[v])) { bPd[v] = od; bPi[v] = oi; }
      }
      float nd = __shfl_xor(bNd[v], off, 32);
      int   ni = __shfl_xor(bNi[v], off, 32);
      if (nd < bNd[v] || (nd == bNd[v] && ni < bNi[v])) { bNd[v] = nd; bNi[v] = ni; }
    }
  }

  __shared__ float sPd[8][16]; __shared__ int sPi[8][16];
  __shared__ float sNd[8][16]; __shared__ int sNi[8][16];
  if (nl == 0) {
#pragma unroll
    for (int v = 0; v < 8; ++v) {
      const int r = v + 8 * h;
      if (PASS == 0) { sPd[w][r] = bPd[v]; sPi[w][r] = bPi[v]; }
      sNd[w][r] = bNd[v]; sNi[w][r] = bNi[v];
    }
  }
  __syncthreads();
  if (tid < 16) {
    float pd = -INF_F; int pi = 0;
    float nd =  INF_F; int ni = 0;
    for (int wv = 0; wv < 8; ++wv) {
      if (PASS == 0) {
        float od = sPd[wv][tid]; int oi = sPi[wv][tid];
        if (od > pd || (od == pd && oi < pi)) { pd = od; pi = oi; }
      }
      float md = sNd[wv][tid]; int mi = sNi[wv][tid];
      if (md < nd || (md == nd && mi < ni)) { nd = md; ni = mi; }
    }
    const int r = rb + tid;
    if (PASS == 0) {
      dp_out[r]   = pd;
      posi_out[r] = pi;
      negh_out[r] = ni;
    } else {
      negsel_out[r] = (nd < 0.5f * INF_F) ? ni : negh_in[r];
    }
  }
  (void)dp_in; (void)negh_in; (void)dp_out; (void)posi_out; (void)negh_out;
  (void)negsel_out;
}

// ---------------------------------------------------------------------------
// kernel 3: per-anchor triplet loss (torch pairwise_distance eps semantics)
// ---------------------------------------------------------------------------
__global__ __launch_bounds__(256) void loss_kernel(
    const float* __restrict__ E, const int* __restrict__ posi,
    const int* __restrict__ negi, float* __restrict__ loss, int N, int D) {
  const int i = blockIdx.x;
  const float* a = E + (size_t)i * D;
  const float* p = E + (size_t)posi[i] * D;
  const float* n = E + (size_t)negi[i] * D;
  float sp = 0.f, sn = 0.f;
  for (int d = threadIdx.x; d < D; d += 256) {
    const float av = a[d];
    const float dp = av - p[d] + EPS_F;
    const float dn = av - n[d] + EPS_F;
    sp += dp * dp;
    sn += dn * dn;
  }
#pragma unroll
  for (int off = 16; off; off >>= 1) {
    sp += __shfl_xor(sp, off, 32);
    sn += __shfl_xor(sn, off, 32);
  }
  __shared__ float wp[8], wn[8];
  if ((threadIdx.x & 31) == 0) {
    wp[threadIdx.x >> 5] = sp;
    wn[threadIdx.x >> 5] = sn;
  }
  __syncthreads();
  if (threadIdx.x == 0) {
    float tp = 0.f, tn = 0.f;
    for (int j = 0; j < 8; ++j) { tp += wp[j]; tn += wn[j]; }
    loss[i] = fmaxf(sqrtf(tp) - sqrtf(tn) + MARGIN_F, 0.f);
  }
}

// kernel 4: deterministic mean -> d_out[0]
__global__ __launch_bounds__(256) void mean_kernel(const float* __restrict__ loss,
                                                   float* __restrict__ out, int N) {
  float s = 0.f;
  for (int i = threadIdx.x; i < N; i += 256) s += loss[i];
#pragma unroll
  for (int off = 16; off; off >>= 1) s += __shfl_xor(s, off, 32);
  __shared__ float p[8];
  if ((threadIdx.x & 31) == 0) p[threadIdx.x >> 5] = s;
  __syncthreads();
  if (threadIdx.x == 0) {
    float t = 0.f;
    for (int j = 0; j < 8; ++j) t += p[j];
    out[0] = t / (float)N;
  }
}

// ---------------------------------------------------------------------------
extern "C" void kernel_launch(void* const* d_in, const int* in_sizes, int n_in,
                              void* d_out, int out_size, void* d_ws, size_t ws_size,
                              hipStream_t stream) {
  (void)n_in; (void)out_size;
  const float* E   = (const float*)d_in[0];
  const int*   tgt = (const int*)d_in[1];
  const int N = in_sizes[1];
  const int D = in_sizes[0] / N;
  const size_t ND = (size_t)N * D;

  // small scratch: 6*N floats (192 KB for N=8192)
  float* ws    = (float*)d_ws;
  float* sq    = ws;                  // [N]
  float* dp    = ws + N;              // [N]
  float* lossv = ws + 2 * (size_t)N;  // [N]
  int*   posi  = (int*)(ws + 3 * (size_t)N);
  int*   negh  = (int*)(ws + 4 * (size_t)N);
  int*   negs  = (int*)(ws + 5 * (size_t)N);

  // fast path: pre-split bf16 hi/lo copies of E (2*ND bf16 = 32 MB, L2-resident)
  const size_t small_bytes = 6 * (size_t)N * sizeof(float);
  const size_t need = small_bytes + 2 * ND * sizeof(__bf16);
  const bool pre = (ws_size >= need);
  __bf16* Ehi = (__bf16*)((char*)d_ws + small_bytes);
  __bf16* Elo = Ehi + ND;

  sq_kernel<<<N, 256, 0, stream>>>(E, sq, N, D);
  if (pre) {
    const int total4 = (int)(ND / 4);
    split_kernel<<<(total4 + 255) / 256, 256, 0, stream>>>(E, Ehi, Elo, total4);
    mine_pass<0, true><<<N / 16, 256, 0, stream>>>(E, Ehi, Elo, tgt, sq, nullptr,
                                                   nullptr, dp, posi, negh,
                                                   nullptr, N, D);
    mine_pass<1, true><<<N / 16, 256, 0, stream>>>(E, Ehi, Elo, tgt, sq, dp, negh,
                                                   nullptr, nullptr, nullptr,
                                                   negs, N, D);
  } else {
    mine_pass<0, false><<<N / 16, 256, 0, stream>>>(E, nullptr, nullptr, tgt, sq,
                                                    nullptr, nullptr, dp, posi,
                                                    negh, nullptr, N, D);
    mine_pass<1, false><<<N / 16, 256, 0, stream>>>(E, nullptr, nullptr, tgt, sq,
                                                    dp, negh, nullptr, nullptr,
                                                    nullptr, negs, N, D);
  }
  loss_kernel<<<N, 256, 0, stream>>>(E, posi, negs, lossv, N, D);
  mean_kernel<<<1, 256, 0, stream>>>(lossv, (float*)d_out, N);
}
